// PulseLTC_67645734912797
// MI455X (gfx1250) — compile-verified, matrix operand order checked by reading, not run on hardware
//
#include <hip/hip_runtime.h>
#include <hip/hip_bf16.h>

// Problem sizes (fixed by the reference)
#define BATCH   64
#define SEQ     256
#define IDIM    128
#define HDIM    128
#define UNFOLDS 6
#define EPSV    1e-8f
#define LOG2E   1.44269504088896340736f

typedef __attribute__((ext_vector_type(16))) _Float16 v16h;
typedef __attribute__((ext_vector_type(8)))  float    v8f;

union V16H { v16h v; uint4 q[2]; };

__device__ __forceinline__ float softplus_f(float x) {
    return (x > 20.0f) ? x : log1pf(__expf(x));
}

// Fast sigmoid of (sig * (v - mu)) given prebaked a = sig*log2e, b = sig*log2e*mu:
//   sigmoid = rcp(1 + exp2(b - a*v));  exp2->inf => rcp->0 (correct saturation).
__device__ __forceinline__ float sig_fast(float a, float b, float v) {
    const float z = fmaf(-a, v, b);
    return __builtin_amdgcn_rcpf(1.0f + __builtin_amdgcn_exp2f(z));
}

// ---------------------------------------------------------------------------
// Precompute: pack per-pair params into float4 tables (exp2-form), softplus
// transforms, fp16 copies of the matvec weights. H*H == I*H == 16384.
//   rec4[e] = ( sigma*log2e, sigma*log2e*mu, softplus(w)*erev, softplus(w) )
//   sen4[e] = ( same for sensory params )
// ---------------------------------------------------------------------------
__global__ __launch_bounds__(256) void ltc_precompute(
    const float* __restrict__ gleak, const float* __restrict__ cm,
    const float* __restrict__ sigma, const float* __restrict__ mu,
    const float* __restrict__ w,     const float* __restrict__ erev,
    const float* __restrict__ s_sigma, const float* __restrict__ s_mu,
    const float* __restrict__ s_w,   const float* __restrict__ s_erev,
    const float* __restrict__ phase_W, const float* __restrict__ sa_W,
    float4* __restrict__ rec4, float4* __restrict__ sen4,
    float* __restrict__ gl, float* __restrict__ cm_t,
    _Float16* __restrict__ pw16, _Float16* __restrict__ sw16)
{
    int e = blockIdx.x * blockDim.x + threadIdx.x;
    if (e < HDIM * HDIM) {
        float wp = softplus_f(w[e]);
        float a  = sigma[e] * LOG2E;
        rec4[e] = make_float4(a, a * mu[e], wp * erev[e], wp);
        float swp = softplus_f(s_w[e]);
        float sa  = s_sigma[e] * LOG2E;
        sen4[e] = make_float4(sa, sa * s_mu[e], swp * s_erev[e], swp);
        pw16[e] = (_Float16)phase_W[e];
        sw16[e] = (_Float16)sa_W[e];
        if (e < HDIM) {
            gl[e]   = softplus_f(gleak[e]);
            cm_t[e] = softplus_f(cm[e]) * (float)UNFOLDS;  // DT = 1.0
        }
    }
}

// ---------------------------------------------------------------------------
// Main LTC scan kernel: one workgroup per batch element, 512 threads (16 waves),
// one WG per WGP. Packed recurrent table (256 KB) pinned in LDS: the 6x-unfold
// loop is one ds_load_b128 per synapse pair (+ 1/4 ds_load_b128 for the state
// broadcast) and ~6 VALU + 2 transcendental ops.
// ---------------------------------------------------------------------------
__global__ __launch_bounds__(512, 1) void ltc_scan(
    const float*  __restrict__ x,        // (B,S,I)
    const float*  __restrict__ h0,       // (B,H)
    const float*  __restrict__ input_w,  // (I)
    const float*  __restrict__ input_b,  // (I)
    const float*  __restrict__ vleak,    // (H)
    const float*  __restrict__ amplitude,// (H)
    const float*  __restrict__ omega,    // (H)
    const float*  __restrict__ phase_b,  // (H)
    const float*  __restrict__ alpha_p,  // scalar
    const float*  __restrict__ beta_p,   // scalar
    const float4* __restrict__ rec4_g,   // (H*H) packed recurrent params
    const float4* __restrict__ sen4_g,   // (I*H) packed sensory params
    const float*  __restrict__ gl_g,     // (H)
    const float*  __restrict__ cmt_g,    // (H)
    const _Float16* __restrict__ pw16,   // (H,H) phase_W fp16
    const _Float16* __restrict__ sw16,   // (H,H) sa_W fp16
    float* __restrict__ out)             // (B,S,H) readout + (B,H) h_final
{
    // LDS: packed recurrent synapse table resident (256 KB) + scan buffers.
    __shared__ __attribute__((aligned(16))) float4 lds_rec[HDIM * HDIM]; // 256 KB
    __shared__ __attribute__((aligned(16))) float sv[HDIM];      // state v
    __shared__ __attribute__((aligned(16))) float sxi[IDIM];     // mapped input
    __shared__ __attribute__((aligned(16))) float snum_s[HDIM];  // sensory numer
    __shared__ __attribute__((aligned(16))) float sden_s[HDIM];  // sensory denom
    __shared__ __attribute__((aligned(16))) float pnum[4][HDIM]; // partials
    __shared__ __attribute__((aligned(16))) float pden[4][HDIM];
    __shared__ __attribute__((aligned(16))) float sphi[HDIM];    // matvec result
    __shared__ __attribute__((aligned(16))) _Float16 svh[HDIM];  // v in fp16
    __shared__ __attribute__((aligned(16))) _Float16 ssh[HDIM];  // sigmoid(v) fp16

    const int b    = blockIdx.x;
    const int tid  = threadIdx.x;
    const int h    = tid & (HDIM - 1);  // output unit this thread reduces into
    const int q    = tid >> 7;          // quarter of the source dim (0..3)
    const int wav  = tid >> 5;
    const int lane = tid & 31;

    // Pin packed recurrent table in LDS (one-time, coalesced b128 copies).
    for (int e = tid; e < HDIM * HDIM; e += 512)
        lds_rec[e] = rec4_g[e];
    if (tid < HDIM) sv[tid] = h0[(size_t)b * HDIM + tid];
    __syncthreads();

    const float alpha = alpha_p[0];
    const float beta  = beta_p[0];

    // Per-h constants for the update threads (tid < H).
    float glh = 0.f, cmth = 0.f, vlh = 0.f, amph = 0.f, omh = 0.f, pbh = 0.f;
    if (tid < HDIM) {
        glh  = gl_g[h];
        cmth = cmt_g[h];
        vlh  = vleak[h];
        amph = amplitude[h];
        omh  = omega[h];
        pbh  = phase_b[h];
    }

    const float* xb = x + (size_t)b * SEQ * IDIM;

    for (int t = 0; t < SEQ; ++t) {
        // ---- input mapping: xi = x_t * input_w + input_b -------------------
        if (tid < IDIM) {
            sxi[tid] = xb[(size_t)t * IDIM + tid] * input_w[tid] + input_b[tid];
            if (t + 1 < SEQ)
                __builtin_prefetch(&xb[(size_t)(t + 1) * IDIM + tid], 0, 0);
        }
        __syncthreads();

        // ---- sensory synapses: once per step (streamed, L2-resident) ------
        {
            float an = 0.f, ad = 0.f;
            const int i0 = q << 5;
            #pragma unroll
            for (int c = 0; c < 8; ++c) {
                const int i = i0 + (c << 2);
                const float4 x4 = *(const float4*)&sxi[i];   // ds_load_b128
                #pragma unroll
                for (int k = 0; k < 4; ++k) {
                    const float xv = (k == 0) ? x4.x : (k == 1) ? x4.y
                                   : (k == 2) ? x4.z : x4.w;
                    const float4 p = sen4_g[(i + k) * HDIM + h]; // global b128
                    const float sg = sig_fast(p.x, p.y, xv);
                    an = fmaf(p.z, sg, an);
                    ad = fmaf(p.w, sg, ad);
                }
            }
            pnum[q][h] = an; pden[q][h] = ad;
        }
        __syncthreads();
        if (tid < HDIM) {
            snum_s[h] = pnum[0][h] + pnum[1][h] + pnum[2][h] + pnum[3][h];
            sden_s[h] = pden[0][h] + pden[1][h] + pden[2][h] + pden[3][h];
        }
        __syncthreads();

        // ---- semi-implicit ODE unfolds (all operands in LDS) --------------
        for (int u = 0; u < UNFOLDS; ++u) {
            float an = 0.f, ad = 0.f;
            const int j0 = q << 5;
            #pragma unroll
            for (int c = 0; c < 8; ++c) {
                const int j = j0 + (c << 2);
                const float4 v4 = *(const float4*)&sv[j];    // ds_load_b128
                #pragma unroll
                for (int k = 0; k < 4; ++k) {
                    const float vv = (k == 0) ? v4.x : (k == 1) ? v4.y
                                   : (k == 2) ? v4.z : v4.w;
                    const float4 p = lds_rec[(j + k) * HDIM + h]; // ds_load_b128
                    const float sg = sig_fast(p.x, p.y, vv);
                    an = fmaf(p.z, sg, an);
                    ad = fmaf(p.w, sg, ad);
                }
            }
            pnum[q][h] = an; pden[q][h] = ad;
            __syncthreads();
            if (tid < HDIM) {
                const float num = pnum[0][h] + pnum[1][h] + pnum[2][h] + pnum[3][h] + snum_s[h];
                const float den = pden[0][h] + pden[1][h] + pden[2][h] + pden[3][h] + sden_s[h];
                const float vv  = sv[h];
                sv[h] = (cmth * vv + glh * vlh + num) *
                        __builtin_amdgcn_rcpf(cmth + glh + den + EPSV);
            }
            __syncthreads();
        }

        // ---- phi = v @ phase_W.T via WMMA (fp16 in, f32 acc) --------------
        if (tid < HDIM) svh[h] = (_Float16)sv[h];
        __syncthreads();
        if (wav < 8) {
            // A: 16x32 tile of phase_W rows; B: v chunk replicated over 16 cols.
            const int m    = lane & 15;
            const int base = (lane >> 4) << 3;  // 0 or 8 (K grouping per lane half)
            const _Float16* rowp = pw16 + (size_t)(wav * 16 + m) * HDIM;
            v8f acc = {0.f, 0.f, 0.f, 0.f, 0.f, 0.f, 0.f, 0.f};
            #pragma unroll
            for (int k0 = 0; k0 < HDIM; k0 += 32) {
                V16H a, bb;
                a.q[0]  = *(const uint4*)(rowp + k0 + base);
                a.q[1]  = *(const uint4*)(rowp + k0 + 16 + base);
                bb.q[0] = *(const uint4*)(&svh[k0 + base]);
                bb.q[1] = *(const uint4*)(&svh[k0 + 16 + base]);
                acc = __builtin_amdgcn_wmma_f32_16x16x32_f16(
                    false, a.v, false, bb.v, (short)0, acc, false, false);
            }
            if ((lane & 15) == 0) {  // lanes 0 (M=0..7) and 16 (M=8..15), column N=0
                const int mbase = wav * 16 + base;
                #pragma unroll
                for (int r = 0; r < 8; ++r) sphi[mbase + r] = acc[r];
            }
        }
        __syncthreads();

        // ---- oscillatory pulse + prepare sigmoid(v) for self-attend -------
        if (tid < HDIM) {
            // |omega*t| <= 2550 rad; result scaled by alpha*amp ~1e-3 ->
            // v_sin_f32 precision is ample.
            float vv = sv[h] + alpha * (amph * __sinf(omh * (float)t + sphi[h] + pbh));
            sv[h] = vv;
            ssh[h] = (_Float16)sig_fast(LOG2E, 0.0f, vv);  // sigmoid(vv)
        }
        __syncthreads();

        // ---- att = sigmoid(v) @ sa_W.T via WMMA ---------------------------
        if (wav < 8) {
            const int m    = lane & 15;
            const int base = (lane >> 4) << 3;
            const _Float16* rowp = sw16 + (size_t)(wav * 16 + m) * HDIM;
            v8f acc = {0.f, 0.f, 0.f, 0.f, 0.f, 0.f, 0.f, 0.f};
            #pragma unroll
            for (int k0 = 0; k0 < HDIM; k0 += 32) {
                V16H a, bb;
                a.q[0]  = *(const uint4*)(rowp + k0 + base);
                a.q[1]  = *(const uint4*)(rowp + k0 + 16 + base);
                bb.q[0] = *(const uint4*)(&ssh[k0 + base]);
                bb.q[1] = *(const uint4*)(&ssh[k0 + 16 + base]);
                acc = __builtin_amdgcn_wmma_f32_16x16x32_f16(
                    false, a.v, false, bb.v, (short)0, acc, false, false);
            }
            if ((lane & 15) == 0) {
                const int mbase = wav * 16 + base;
                #pragma unroll
                for (int r = 0; r < 8; ++r) sphi[mbase + r] = acc[r];
            }
        }
        __syncthreads();

        // ---- state update + write readout ---------------------------------
        if (tid < HDIM) {
            const float vv = sv[h] + beta * sphi[h];
            sv[h] = vv;
            out[((size_t)b * SEQ + t) * HDIM + h] = vv;
        }
        __syncthreads();
    }

    // h_final appended after the (B,S,H) readout
    if (tid < HDIM)
        out[(size_t)BATCH * SEQ * HDIM + (size_t)b * HDIM + h] = sv[h];
}

// ---------------------------------------------------------------------------
extern "C" void kernel_launch(void* const* d_in, const int* in_sizes, int n_in,
                              void* d_out, int out_size, void* d_ws, size_t ws_size,
                              hipStream_t stream) {
    (void)in_sizes; (void)n_in; (void)out_size; (void)ws_size;

    const float* x        = (const float*)d_in[0];
    const float* h0       = (const float*)d_in[1];
    const float* input_w  = (const float*)d_in[2];
    const float* input_b  = (const float*)d_in[3];
    const float* gleak    = (const float*)d_in[4];
    const float* vleak    = (const float*)d_in[5];
    const float* cm       = (const float*)d_in[6];
    const float* sigma    = (const float*)d_in[7];
    const float* mu       = (const float*)d_in[8];
    const float* w        = (const float*)d_in[9];
    const float* erev     = (const float*)d_in[10];
    const float* s_sigma  = (const float*)d_in[11];
    const float* s_mu     = (const float*)d_in[12];
    const float* s_w      = (const float*)d_in[13];
    const float* s_erev   = (const float*)d_in[14];
    const float* amplitude= (const float*)d_in[15];
    const float* omega    = (const float*)d_in[16];
    const float* phase_W  = (const float*)d_in[17];
    const float* phase_b  = (const float*)d_in[18];
    const float* alpha    = (const float*)d_in[19];
    const float* sa_W     = (const float*)d_in[20];
    const float* beta     = (const float*)d_in[21];

    // Workspace layout: two packed float4 tables, scalars, fp16 weights (~578 KB).
    float4* rec4   = (float4*)d_ws;                      // 16384 float4 (256 KB)
    float4* sen4   = rec4 + HDIM * HDIM;                 // 16384 float4 (256 KB)
    float*  gl     = (float*)(sen4 + IDIM * HDIM);       // 128
    float*  cm_t   = gl + HDIM;                          // 128
    _Float16* pw16 = (_Float16*)(cm_t + HDIM);           // 16384 halves
    _Float16* sw16 = pw16 + HDIM * HDIM;                 // 16384 halves

    ltc_precompute<<<(HDIM * HDIM + 255) / 256, 256, 0, stream>>>(
        gleak, cm, sigma, mu, w, erev, s_sigma, s_mu, s_w, s_erev,
        phase_W, sa_W, rec4, sen4, gl, cm_t, pw16, sw16);

    ltc_scan<<<BATCH, 512, 0, stream>>>(
        x, h0, input_w, input_b, vleak, amplitude, omega, phase_b,
        alpha, beta, rec4, sen4, gl, cm_t, pw16, sw16,
        (float*)d_out);
}
